// TopKRouter_57578331570799
// MI455X (gfx1250) — matrix operand order repaired and checked by reading, not compiled, not run on hardware
//
#include <hip/hip_runtime.h>
#include <hip/hip_bf16.h>

#define T_TOKENS 16384
#define D_MODEL  2048
#define N_EXP    64
#define B_STRIDE 2056   // padded LDS row stride (bf16 elems): 4112B -> conflict-free banks

typedef __attribute__((ext_vector_type(16))) __bf16 v16bf;
typedef __attribute__((ext_vector_type(8)))  float  v8f;
typedef __attribute__((ext_vector_type(8)))  float  f32x8;

__device__ __forceinline__ bool pair_gt(float va, int ia, float vb, int ib) {
    // "greater" with jax top_k tie-break: equal values -> lower index wins
    return (va > vb) || (va == vb && ia < ib);
}

// Kernel 1: convert router weights f32 -> bf16 into workspace; zero the
// 128-float global accumulator region (probsum[64] + count[64]).
__global__ void prep_kernel(const float* __restrict__ w,
                            __bf16* __restrict__ wbf,
                            float* __restrict__ gacc) {
    int i = blockIdx.x * blockDim.x + threadIdx.x;
    if (i < N_EXP * D_MODEL) wbf[i] = (__bf16)w[i];
    if (blockIdx.x == 0 && threadIdx.x < 128) gacc[threadIdx.x] = 0.0f;
}

// Kernel 2: WMMA router GEMM (B staged in LDS, A + B software-pipelined with a
// hard scheduling fence) + noise + logits + top-2 + softmax stats.
__global__ void __launch_bounds__(256, 1)
router_kernel(const float* __restrict__ hidden,
              const __bf16* __restrict__ wbf,
              const float* __restrict__ noise,
              float* __restrict__ out,
              float* __restrict__ gacc) {
    __shared__ float s_prob[N_EXP];
    __shared__ float s_cnt[N_EXP];
    extern __shared__ char smem_raw[];
    __bf16* sB = (__bf16*)smem_raw;      // [64][B_STRIDE] padded bf16 weights

    const int tid = threadIdx.x;
    if (tid < 64)       s_prob[tid] = 0.0f;
    else if (tid < 128) s_cnt[tid - 64] = 0.0f;

    // ---- Stage full bf16 weight matrix into LDS (256 KB, shared by 8 waves) ----
    {
        const uint4* gB = (const uint4*)wbf;             // 16B = 8 bf16 per chunk
        for (int c = tid; c < N_EXP * (D_MODEL / 8); c += 256) {
            int e = c >> 8;                              // chunk / 256
            int o = c & 255;                             // chunk within expert row
            *(uint4*)(sB + (size_t)e * B_STRIDE + o * 8) = gB[(size_t)e * 256 + o];
        }
    }
    __syncthreads();

    const int lane = tid & 31;
    const int wave = tid >> 5;
    const int lrow = lane & 15;   // row (A) / column-expert (B/D) within tile
    const int hi   = lane >> 4;   // half-wave selector
    const int t0   = blockIdx.x * 128 + wave * 16;

    v8f acc0 = {}, acc1 = {}, acc2 = {}, acc3 = {};

    // A: lane holds row t0+lrow, K segment koff..koff+15 (contiguous), koff = hi?16:0
    const float* arow = hidden + (size_t)(t0 + lrow) * D_MODEL + (hi ? 16 : 0);
    // B (LDS): lane holds expert lrow + 16*tile, K offset (hi?16:0), 16 contiguous bf16
    const __bf16* sB0 = sB + (size_t)lrow * B_STRIDE + (hi ? 16 : 0);
    const __bf16* sB1 = sB0 + 16 * B_STRIDE;
    const __bf16* sB2 = sB0 + 32 * B_STRIDE;
    const __bf16* sB3 = sB0 + 48 * B_STRIDE;

    // Software pipeline: A (global NT) and B (LDS) both prefetched one k-step ahead.
    f32x8 sA0 = __builtin_nontemporal_load((const f32x8*)(arow));
    f32x8 sA1 = __builtin_nontemporal_load((const f32x8*)(arow + 8));
    v16bf b0 = *(const v16bf*)(sB0);
    v16bf b1 = *(const v16bf*)(sB1);
    v16bf b2 = *(const v16bf*)(sB2);
    v16bf b3 = *(const v16bf*)(sB3);

#pragma unroll 2
    for (int k0 = 0; k0 < D_MODEL; k0 += 32) {
        const int kn = (k0 + 32) & (D_MODEL - 1);   // wraps to 0 on last iter (always valid)

        // ---- issue next iteration's loads (must stay above the fence) ----
        f32x8 nA0 = __builtin_nontemporal_load((const f32x8*)(arow + kn));
        f32x8 nA1 = __builtin_nontemporal_load((const f32x8*)(arow + kn + 8));
        v16bf nb0 = *(const v16bf*)(sB0 + kn);
        v16bf nb1 = *(const v16bf*)(sB1 + kn);
        v16bf nb2 = *(const v16bf*)(sB2 + kn);
        v16bf nb3 = *(const v16bf*)(sB3 + kn);

        v16bf a;
#pragma unroll
        for (int i = 0; i < 8; ++i) { a[i] = (__bf16)sA0[i]; a[8 + i] = (__bf16)sA1[i]; }

        // Hard scheduling fence: loads above may not sink below, WMMAs below may
        // not hoist above -> WMMAs consume fragments loaded one full k-step ago,
        // so their dscnt/loadcnt waits are already satisfied.
        __builtin_amdgcn_sched_barrier(0);

        acc0 = __builtin_amdgcn_wmma_f32_16x16x32_bf16(false, a, false, b0, (short)0, acc0, false, false);
        acc1 = __builtin_amdgcn_wmma_f32_16x16x32_bf16(false, a, false, b1, (short)0, acc1, false, false);
        acc2 = __builtin_amdgcn_wmma_f32_16x16x32_bf16(false, a, false, b2, (short)0, acc2, false, false);
        acc3 = __builtin_amdgcn_wmma_f32_16x16x32_bf16(false, a, false, b3, (short)0, acc3, false, false);

        sA0 = nA0; sA1 = nA1;
        b0 = nb0; b1 = nb1; b2 = nb2; b3 = nb3;
    }

    float* out_idx = out;              // [16384, 2] indices (as float)
    float* out_w   = out + 32768;      // [16384, 2] softmaxed top-2 weights
    float* out_log = out + 65536;      // [16384, 64] router logits

#pragma unroll
    for (int j = 0; j < 8; ++j) {
        const int t = t0 + j + 8 * hi;               // token for D row j in this half
        const size_t nb = (size_t)t * N_EXP + lrow;

        float lv0 = acc0[j] + 0.1f * __builtin_nontemporal_load(&noise[nb]);
        float lv1 = acc1[j] + 0.1f * __builtin_nontemporal_load(&noise[nb + 16]);
        float lv2 = acc2[j] + 0.1f * __builtin_nontemporal_load(&noise[nb + 32]);
        float lv3 = acc3[j] + 0.1f * __builtin_nontemporal_load(&noise[nb + 48]);

        __builtin_nontemporal_store(lv0, &out_log[nb]);
        __builtin_nontemporal_store(lv1, &out_log[nb + 16]);
        __builtin_nontemporal_store(lv2, &out_log[nb + 32]);
        __builtin_nontemporal_store(lv3, &out_log[nb + 48]);

        // local top-2 among this lane's 4 experts
        float v1 = lv0, v2 = lv1; int i1 = lrow, i2 = lrow + 16;
        if (pair_gt(v2, i2, v1, i1)) { float tv = v1; int ti = i1; v1 = v2; i1 = i2; v2 = tv; i2 = ti; }
        if (pair_gt(lv2, lrow + 32, v1, i1)) { v2 = v1; i2 = i1; v1 = lv2; i1 = lrow + 32; }
        else if (pair_gt(lv2, lrow + 32, v2, i2)) { v2 = lv2; i2 = lrow + 32; }
        if (pair_gt(lv3, lrow + 48, v1, i1)) { v2 = v1; i2 = i1; v1 = lv3; i1 = lrow + 48; }
        else if (pair_gt(lv3, lrow + 48, v2, i2)) { v2 = lv3; i2 = lrow + 48; }

        // butterfly merge of sorted top-2 pairs across the 16 lanes of this half
#pragma unroll
        for (int m = 1; m < 16; m <<= 1) {
            float ov1 = __shfl_xor(v1, m, 16);
            int   oi1 = __shfl_xor(i1, m, 16);
            float ov2 = __shfl_xor(v2, m, 16);
            int   oi2 = __shfl_xor(i2, m, 16);
            if (pair_gt(ov1, oi1, v1, i1)) {
                if (pair_gt(v1, i1, ov2, oi2)) { v2 = v1; i2 = i1; }
                else                           { v2 = ov2; i2 = oi2; }
                v1 = ov1; i1 = oi1;
            } else if (pair_gt(ov1, oi1, v2, i2)) {
                v2 = ov1; i2 = oi1;
            }
        }

        // full 64-way softmax denominator (global max == v1 after reduction)
        const float mx = v1;
        float s = __expf(lv0 - mx) + __expf(lv1 - mx) + __expf(lv2 - mx) + __expf(lv3 - mx);
#pragma unroll
        for (int m = 1; m < 16; m <<= 1) s += __shfl_xor(s, m, 16);
        const float inv = 1.0f / s;

        atomicAdd(&s_prob[lrow],      __expf(lv0 - mx) * inv);
        atomicAdd(&s_prob[lrow + 16], __expf(lv1 - mx) * inv);
        atomicAdd(&s_prob[lrow + 32], __expf(lv2 - mx) * inv);
        atomicAdd(&s_prob[lrow + 48], __expf(lv3 - mx) * inv);

        if (lrow == j) {   // one writer per token
            out_idx[t * 2 + 0] = (float)i1;
            out_idx[t * 2 + 1] = (float)i2;
            float w1 = 1.0f / (1.0f + __expf(v2 - v1));  // softmax over [v1, v2]
            out_w[t * 2 + 0] = w1;
            out_w[t * 2 + 1] = 1.0f - w1;
            atomicAdd(&s_cnt[i1], 1.0f);
            atomicAdd(&s_cnt[i2], 1.0f);
        }
    }

    __syncthreads();
    if (tid < 64)       atomicAdd(&gacc[tid], s_prob[tid]);
    else if (tid < 128) atomicAdd(&gacc[tid], s_cnt[tid - 64]);
}

// Kernel 3: aux load-balancing loss scalar.
__global__ void aux_kernel(const float* __restrict__ gacc, float* __restrict__ out) {
    int l = threadIdx.x;  // 32 threads, one wave
    float s = 0.0f;
    for (int e = l; e < N_EXP; e += 32) {
        float usage = gacc[64 + e] * (1.0f / (16384.0f * 2.0f));
        float meanp = gacc[e]      * (1.0f / 16384.0f);
        s += usage * meanp;
    }
#pragma unroll
    for (int m = 16; m >= 1; m >>= 1) s += __shfl_xor(s, m, 32);
    if (l == 0) out[1114112] = s * 64.0f * 0.001f;
}

extern "C" void kernel_launch(void* const* d_in, const int* in_sizes, int n_in,
                              void* d_out, int out_size, void* d_ws, size_t ws_size,
                              hipStream_t stream) {
    const float* hidden = (const float*)d_in[0];
    const float* weight = (const float*)d_in[1];
    const float* noise  = (const float*)d_in[2];
    float* out  = (float*)d_out;
    float* gacc = (float*)d_ws;                         // 128 floats
    __bf16* wbf = (__bf16*)((char*)d_ws + 512);         // 256 KB bf16 weights

    const size_t dyn_lds = (size_t)N_EXP * B_STRIDE * sizeof(__bf16);  // ~257 KB

    prep_kernel<<<(N_EXP * D_MODEL + 255) / 256, 256, 0, stream>>>(weight, wbf, gacc);
    router_kernel<<<T_TOKENS / 128, 256, dyn_lds, stream>>>(hidden, wbf, noise, out, gacc);
    aux_kernel<<<1, 32, 0, stream>>>(gacc, out);
}
